// BiMPM_NN_69269232550469
// MI455X (gfx1250) — compile-verified, matrix operand order checked by reading, not run on hardware
//
#include <hip/hip_runtime.h>
#include <math.h>

// ---------------------------------------------------------------------------
// Types for CDNA5 WMMA
// ---------------------------------------------------------------------------
typedef __attribute__((ext_vector_type(16))) _Float16 v16h;
typedef __attribute__((ext_vector_type(8)))  _Float16 v8h;
typedef __attribute__((ext_vector_type(8)))  float    v8f;

#define B_   128
#define S_   64
#define W_   16
#define P_   20
#define H_   100
#define CH_  50
#define CE_  20
#define WE_  300

// char LSTM tile geometry
#define C_KXP 32      // pad(20)
#define C_KP  96      // pad(20)+pad(50)=32+64
#define C_NP  208     // pad(200)
// context LSTM
#define X_KXP 352     // pad(350) = 11*32
#define X_KP  480     // 352 + 128
#define X_NP  400     // 25*16
// aggregation LSTM
#define A_KXP 64      // pad(40)
#define A_KP  192     // 64 + 128
#define A_NP  400

__device__ __forceinline__ float sigf(float x) { return 1.0f / (1.0f + __expf(-x)); }

// Load A fragment (16x32 f16) from LDS tile. Row m = lane%16.
// element j: K = k0 + (j>=8 ? 16 : 0) + (lane>=16 ? 8 : 0) + (j&7)
__device__ __forceinline__ v16h load_a_frag(const _Float16* Alds, int ldA, int k0) {
  int lane = threadIdx.x & 31;
  const _Float16* p = Alds + (lane & 15) * ldA + k0 + ((lane >> 4) << 3);
  v8h lo = *(const v8h*)(p);
  v8h hi = *(const v8h*)(p + 16);
  return __builtin_shufflevector(lo, hi, 0, 1, 2, 3, 4, 5, 6, 7,
                                 8, 9, 10, 11, 12, 13, 14, 15);
}

// Load B fragment from fragment-packed weights: contiguous 32B per lane.
__device__ __forceinline__ v16h load_b_frag(const _Float16* Wp, int nt_n, int kt, int nt) {
  int lane = threadIdx.x & 31;
  const _Float16* p = Wp + (((size_t)(kt * nt_n + nt) * 32 + lane) << 4);
  return *(const v16h*)p;
}

// Scatter f32 accumulator tile to LDS: N = nt*16 + lane%16, M = r + 8*(lane>=16)
__device__ __forceinline__ void store_acc(float* Glds, int ldG, int nt, v8f acc) {
  int lane = threadIdx.x & 31;
  int n = nt * 16 + (lane & 15);
  int mb = (lane >> 4) << 3;
#pragma unroll
  for (int r = 0; r < 8; ++r) Glds[(mb + r) * ldG + n] = acc[r];
}

// ---------------------------------------------------------------------------
// Weight repack: fp32 (Wih: 4H x Din, Whh: 4H x Hh) -> fp16 fragment-packed
// combined B matrix of shape KPad x NPad, K = [x-part | h-part] zero padded.
// ---------------------------------------------------------------------------
__global__ void pack_w_kernel(const float* __restrict__ Wih,
                              const float* __restrict__ Whh,
                              _Float16* __restrict__ Wp,
                              int Din, int Hh, int KxPad, int KPad, int NPad) {
  int o = blockIdx.x * blockDim.x + threadIdx.x;
  int total = KPad * NPad;
  if (o >= total) return;
  int j    = o & 15;
  int lane = (o >> 4) & 31;
  int tile = o >> 9;
  int NT = NPad >> 4;
  int nt = tile % NT;
  int kt = tile / NT;
  int n = nt * 16 + (lane & 15);
  int k = kt * 32 + ((j >> 3) << 4) + ((lane >> 4) << 3) + (j & 7);
  float v = 0.0f;
  if (n < 4 * Hh) {
    if (k < Din)                              v = Wih[n * Din + k];
    else if (k >= KxPad && k < KxPad + Hh)    v = Whh[n * Hh + (k - KxPad)];
  }
  Wp[o] = (_Float16)v;
}

// ---------------------------------------------------------------------------
// Char LSTM: 16 words per block, 16 timesteps, h/c resident in LDS.
// Per step: gather char embeddings, WMMA gates (13 N-tiles x 3 K-steps), cell.
// ---------------------------------------------------------------------------
__global__ __launch_bounds__(256) void char_lstm_kernel(
    const float* __restrict__ char_emb, const int* __restrict__ chars,
    const _Float16* __restrict__ Wp, const float* __restrict__ bias,
    float* __restrict__ Hout) {
  __shared__ __attribute__((aligned(32))) _Float16 Alds[16 * C_KP];
  __shared__ float Glds[16 * C_NP];
  __shared__ float Clds[16 * CH_];
  const int tid = threadIdx.x;
  const int w0 = blockIdx.x * 16;
  const int lane = tid & 31;
  const int wave = tid >> 5;

  for (int i = tid; i < 16 * C_KP; i += 256) Alds[i] = (_Float16)0.0f;
  for (int i = tid; i < 16 * CH_; i += 256) Clds[i] = 0.0f;
  __syncthreads();

  for (int t = 0; t < W_; ++t) {
    // stage char embeddings for this step
    for (int i = tid; i < 16 * CE_; i += 256) {
      int m = i / CE_, k = i % CE_;
      int ch = chars[(w0 + m) * W_ + t];
      Alds[m * C_KP + k] = (_Float16)char_emb[ch * CE_ + k];
    }
    __syncthreads();
    // gates = [x h] @ Wcomb
    for (int nt = wave; nt < (C_NP >> 4); nt += 8) {
      if (nt + 8 < (C_NP >> 4))
        __builtin_prefetch(Wp + (((size_t)(nt + 8) * 32 + lane) << 4), 0, 3);
      v8f acc = {};
#pragma unroll
      for (int kt = 0; kt < (C_KP >> 5); ++kt) {
        v16h a = load_a_frag(Alds, C_KP, kt << 5);
        v16h b = load_b_frag(Wp, C_NP >> 4, kt, nt);
        acc = __builtin_amdgcn_wmma_f32_16x16x32_f16(false, a, false, b,
                                                     (short)0, acc, false, false);
      }
      store_acc(Glds, C_NP, nt, acc);
    }
    __syncthreads();
    // cell update (div-free mapping: m = tid&15, n = (tid>>4) + 16*k)
    {
      int m = tid & 15;
      const float* g = Glds + m * C_NP;
      for (int n = (tid >> 4); n < CH_; n += 16) {
        float gi = g[n]            + bias[n];
        float gf = g[CH_ + n]      + bias[CH_ + n];
        float gg = g[2 * CH_ + n]  + bias[2 * CH_ + n];
        float go = g[3 * CH_ + n]  + bias[3 * CH_ + n];
        int e = m * CH_ + n;
        float c = sigf(gf) * Clds[e] + sigf(gi) * tanhf(gg);
        float h = sigf(go) * tanhf(c);
        Clds[e] = c;
        Alds[m * C_KP + C_KXP + n] = (_Float16)h;
        if (t == W_ - 1) Hout[(size_t)(w0 + m) * CH_ + n] = h;
      }
    }
    __syncthreads();
  }
}

// ---------------------------------------------------------------------------
// Build fp16 context input X: [word_emb(300) | charH(50) | pad(2)]
// ---------------------------------------------------------------------------
__global__ void build_x_kernel(const float* __restrict__ word_emb,
                               const int* __restrict__ words,
                               const float* __restrict__ charH,
                               _Float16* __restrict__ X) {
  int i = blockIdx.x * blockDim.x + threadIdx.x;
  int total = B_ * S_ * X_KXP;
  if (i >= total) return;
  int k = i % X_KXP;
  int ws = i / X_KXP;
  float v = 0.0f;
  if (k < WE_)            v = word_emb[(size_t)words[ws] * WE_ + k];
  else if (k < WE_ + CH_) v = charH[(size_t)ws * CH_ + (k - WE_)];
  X[i] = (_Float16)v;
}

// ---------------------------------------------------------------------------
// Generic persistent LSTM (H=100): 16 batch rows per block, T=64 steps,
// 4 instances selected by blockIdx.y. Compile-time K geometry.
// ---------------------------------------------------------------------------
struct LstmInst {
  const _Float16* X;    // (B_, T, KXP) f16
  const _Float16* Wp;   // fragment-packed (KP x 400) f16
  const float*    bias; // 4*H
  float*          hs;   // (B_, T, H) f32 or null
  float*          hlast;// (B_, H)    f32 or null
  int             reverse;
};
struct LstmArgs {
  LstmInst inst[4];
};

template <int KXP, int KP>
__global__ __launch_bounds__(256) void lstm_kernel(LstmArgs args) {
  __shared__ __attribute__((aligned(32))) _Float16 Alds[16 * KP];
  __shared__ float Glds[16 * X_NP];
  __shared__ float Clds[16 * H_];
  const LstmInst inst = args.inst[blockIdx.y];
  const int tid = threadIdx.x;
  const int row0 = blockIdx.x * 16;
  const int lane = tid & 31;
  const int wave = tid >> 5;
  constexpr int KT = KP >> 5;
  constexpr int NT = X_NP >> 4;
  constexpr int CHUNKS = KXP / 8;     // 16B chunks per row

  for (int i = tid; i < 16 * KP; i += 256) Alds[i] = (_Float16)0.0f;
  for (int i = tid; i < 16 * H_; i += 256) Clds[i] = 0.0f;
  __syncthreads();

  for (int it = 0; it < S_; ++it) {
    const int t = inst.reverse ? (S_ - 1 - it) : it;
    // stage x(t) for the 16 rows of this block (16B vector copies)
    for (int i = tid; i < 16 * CHUNKS; i += 256) {
      int m = i / CHUNKS, c = i % CHUNKS;
      const uint4* src =
          (const uint4*)(inst.X + (((size_t)(row0 + m) * S_) + t) * KXP) + c;
      *((uint4*)(Alds + m * KP) + c) = *src;
    }
    __syncthreads();
    // gates = [x h] @ Wcomb : 25 N-tiles x KT K-steps of WMMA
    for (int nt = wave; nt < NT; nt += 8) {
      if (nt + 8 < NT)
        __builtin_prefetch(inst.Wp + (((size_t)(nt + 8) * 32 + lane) << 4), 0, 3);
      v8f acc = {};
#pragma unroll
      for (int kt = 0; kt < KT; ++kt) {
        v16h a = load_a_frag(Alds, KP, kt << 5);
        v16h b = load_b_frag(inst.Wp, NT, kt, nt);
        acc = __builtin_amdgcn_wmma_f32_16x16x32_f16(false, a, false, b,
                                                     (short)0, acc, false, false);
      }
      store_acc(Glds, X_NP, nt, acc);
    }
    __syncthreads();
    // LSTM cell (fp32), div-free mapping
    {
      int m = tid & 15;
      const float* g = Glds + m * X_NP;
      for (int n = (tid >> 4); n < H_; n += 16) {
        float gi = g[n]           + inst.bias[n];
        float gf = g[H_ + n]      + inst.bias[H_ + n];
        float gg = g[2 * H_ + n]  + inst.bias[2 * H_ + n];
        float go = g[3 * H_ + n]  + inst.bias[3 * H_ + n];
        int e = m * H_ + n;
        float c = sigf(gf) * Clds[e] + sigf(gi) * tanhf(gg);
        float h = sigf(go) * tanhf(c);
        Clds[e] = c;
        Alds[m * KP + KXP + n] = (_Float16)h;
        if (inst.hs)
          inst.hs[(((size_t)(row0 + m) * S_) + t) * H_ + n] = h;
        if (inst.hlast && it == S_ - 1)
          inst.hlast[(size_t)(row0 + m) * H_ + n] = h;
      }
    }
    __syncthreads();
  }
}

// ---------------------------------------------------------------------------
// Multi-perspective full matching -> fp16 padded agg inputs (B,S,64)
// blockIdx.y==0 -> m1 (s1 vs s2-final), ==1 -> m2
// ---------------------------------------------------------------------------
__global__ __launch_bounds__(64) void match_kernel(
    const float* __restrict__ hs1f, const float* __restrict__ hs1b,
    const float* __restrict__ hs2f, const float* __restrict__ hs2b,
    const float* __restrict__ w_fw, const float* __restrict__ w_bw,
    _Float16* __restrict__ Xagg1, _Float16* __restrict__ Xagg2) {
  __shared__ float hf[H_], hlf[H_], hb[H_], hlb[H_];
  const int bs = blockIdx.x;
  const int b = bs >> 6, s = bs & 63;
  const int which = blockIdx.y;
  const float *af, *bf, *ab, *bb;
  if (which == 0) { af = hs1f; bf = hs2f; ab = hs1b; bb = hs2b; }
  else            { af = hs2f; bf = hs1f; ab = hs2b; bb = hs1b; }
  const size_t rT = (size_t)b * S_;
  for (int i = threadIdx.x; i < H_; i += 64) {
    hf[i]  = af[(rT + s) * H_ + i];
    hlf[i] = bf[(rT + (S_ - 1)) * H_ + i];
    hb[i]  = ab[(rT + s) * H_ + i];
    hlb[i] = bb[(rT + 0) * H_ + i];
  }
  __syncthreads();
  const int tid = threadIdx.x;
  float val = 0.0f;
  if (tid < 2 * P_) {
    const float* w  = (tid < P_) ? (w_fw + tid * H_) : (w_bw + (tid - P_) * H_);
    const float* h1 = (tid < P_) ? hf : hb;
    const float* h2 = (tid < P_) ? hlf : hlb;
    float dot = 0.0f, na = 0.0f, nb = 0.0f;
    for (int h = 0; h < H_; ++h) {
      float a = h1[h] * w[h];
      float c = h2[h] * w[h];
      dot += a * c; na += a * a; nb += c * c;
    }
    na = sqrtf(na); nb = sqrtf(nb);
    val = dot / (fmaxf(na, 1e-8f) * fmaxf(nb, 1e-8f));
  }
  _Float16* out = (which == 0) ? Xagg1 : Xagg2;
  out[(size_t)bs * A_KXP + tid] = (_Float16)val;
}

// ---------------------------------------------------------------------------
// Final linear: feat(128,400) @ W(400,3) + b -> out(128,3)
// ---------------------------------------------------------------------------
__global__ void pred_kernel(const float* __restrict__ h1f, const float* __restrict__ h1b,
                            const float* __restrict__ h2f, const float* __restrict__ h2b,
                            const float* __restrict__ Wm, const float* __restrict__ bv,
                            float* __restrict__ out) {
  int i = blockIdx.x * blockDim.x + threadIdx.x;
  if (i >= B_ * 3) return;
  int b = i / 3, j = i % 3;
  float acc = bv[j];
  for (int k = 0; k < H_; ++k) acc += h1f[b * H_ + k] * Wm[(k)        * 3 + j];
  for (int k = 0; k < H_; ++k) acc += h1b[b * H_ + k] * Wm[(H_ + k)   * 3 + j];
  for (int k = 0; k < H_; ++k) acc += h2f[b * H_ + k] * Wm[(2*H_ + k) * 3 + j];
  for (int k = 0; k < H_; ++k) acc += h2b[b * H_ + k] * Wm[(3*H_ + k) * 3 + j];
  out[i] = acc;
}

// ---------------------------------------------------------------------------
// Host launch
// ---------------------------------------------------------------------------
extern "C" void kernel_launch(void* const* d_in, const int* in_sizes, int n_in,
                              void* d_out, int out_size, void* d_ws, size_t ws_size,
                              hipStream_t stream) {
  (void)in_sizes; (void)n_in; (void)out_size; (void)ws_size;
  const float* char_emb = (const float*)d_in[0];
  const float* word_emb = (const float*)d_in[1];
  const float* char_Wih = (const float*)d_in[2];
  const float* char_Whh = (const float*)d_in[3];
  const float* char_b   = (const float*)d_in[4];
  const float* ctxf_Wih = (const float*)d_in[5];
  const float* ctxf_Whh = (const float*)d_in[6];
  const float* ctxf_b   = (const float*)d_in[7];
  const float* ctxb_Wih = (const float*)d_in[8];
  const float* ctxb_Whh = (const float*)d_in[9];
  const float* ctxb_b   = (const float*)d_in[10];
  const float* aggf_Wih = (const float*)d_in[11];
  const float* aggf_Whh = (const float*)d_in[12];
  const float* aggf_b   = (const float*)d_in[13];
  const float* aggb_Wih = (const float*)d_in[14];
  const float* aggb_Whh = (const float*)d_in[15];
  const float* aggb_b   = (const float*)d_in[16];
  const float* w_fw     = (const float*)d_in[17];
  const float* w_bw     = (const float*)d_in[18];
  const float* pred_W   = (const float*)d_in[19];
  const float* pred_b   = (const float*)d_in[20];
  const int*   s1_chars = (const int*)d_in[21];
  const int*   s2_chars = (const int*)d_in[22];
  const int*   s1_words = (const int*)d_in[23];
  const int*   s2_words = (const int*)d_in[24];
  float* out = (float*)d_out;

  // workspace layout
  char* base = (char*)d_ws;
  size_t off = 0;
  auto alloc = [&](size_t bytes) -> char* {
    char* p = base + off;
    off += (bytes + 255) & ~(size_t)255;
    return p;
  };
  _Float16* WPchar = (_Float16*)alloc((size_t)C_KP * C_NP * 2);
  _Float16* WPctxf = (_Float16*)alloc((size_t)X_KP * X_NP * 2);
  _Float16* WPctxb = (_Float16*)alloc((size_t)X_KP * X_NP * 2);
  _Float16* WPaggf = (_Float16*)alloc((size_t)A_KP * A_NP * 2);
  _Float16* WPaggb = (_Float16*)alloc((size_t)A_KP * A_NP * 2);
  float* charH1 = (float*)alloc((size_t)B_ * S_ * CH_ * 4);
  float* charH2 = (float*)alloc((size_t)B_ * S_ * CH_ * 4);
  _Float16* X1 = (_Float16*)alloc((size_t)B_ * S_ * X_KXP * 2);
  _Float16* X2 = (_Float16*)alloc((size_t)B_ * S_ * X_KXP * 2);
  float* HSs1f = (float*)alloc((size_t)B_ * S_ * H_ * 4);
  float* HSs1b = (float*)alloc((size_t)B_ * S_ * H_ * 4);
  float* HSs2f = (float*)alloc((size_t)B_ * S_ * H_ * 4);
  float* HSs2b = (float*)alloc((size_t)B_ * S_ * H_ * 4);
  _Float16* Xagg1 = (_Float16*)alloc((size_t)B_ * S_ * A_KXP * 2);
  _Float16* Xagg2 = (_Float16*)alloc((size_t)B_ * S_ * A_KXP * 2);
  float* h1f = (float*)alloc((size_t)B_ * H_ * 4);
  float* h1b = (float*)alloc((size_t)B_ * H_ * 4);
  float* h2f = (float*)alloc((size_t)B_ * H_ * 4);
  float* h2b = (float*)alloc((size_t)B_ * H_ * 4);

  // 1) pack weights (fp32 -> fragment-ordered fp16)
  auto packGrid = [](int total) { return dim3((total + 255) / 256); };
  pack_w_kernel<<<packGrid(C_KP * C_NP), 256, 0, stream>>>(char_Wih, char_Whh, WPchar,
                                                           CE_, CH_, C_KXP, C_KP, C_NP);
  pack_w_kernel<<<packGrid(X_KP * X_NP), 256, 0, stream>>>(ctxf_Wih, ctxf_Whh, WPctxf,
                                                           WE_ + CH_, H_, X_KXP, X_KP, X_NP);
  pack_w_kernel<<<packGrid(X_KP * X_NP), 256, 0, stream>>>(ctxb_Wih, ctxb_Whh, WPctxb,
                                                           WE_ + CH_, H_, X_KXP, X_KP, X_NP);
  pack_w_kernel<<<packGrid(A_KP * A_NP), 256, 0, stream>>>(aggf_Wih, aggf_Whh, WPaggf,
                                                           2 * P_, H_, A_KXP, A_KP, A_NP);
  pack_w_kernel<<<packGrid(A_KP * A_NP), 256, 0, stream>>>(aggb_Wih, aggb_Whh, WPaggb,
                                                           2 * P_, H_, A_KXP, A_KP, A_NP);

  // 2) char LSTMs (8192 words each -> 512 blocks of 16 words)
  char_lstm_kernel<<<dim3((B_ * S_) / 16), 256, 0, stream>>>(char_emb, s1_chars, WPchar,
                                                             char_b, charH1);
  char_lstm_kernel<<<dim3((B_ * S_) / 16), 256, 0, stream>>>(char_emb, s2_chars, WPchar,
                                                             char_b, charH2);

  // 3) build fp16 context inputs
  {
    int total = B_ * S_ * X_KXP;
    build_x_kernel<<<packGrid(total), 256, 0, stream>>>(word_emb, s1_words, charH1, X1);
    build_x_kernel<<<packGrid(total), 256, 0, stream>>>(word_emb, s2_words, charH2, X2);
  }

  // 4) context BiLSTMs: 4 instances in one launch (grid.y)
  {
    LstmArgs a;
    a.inst[0] = { X1, WPctxf, ctxf_b, HSs1f, nullptr, 0 };
    a.inst[1] = { X1, WPctxb, ctxb_b, HSs1b, nullptr, 1 };
    a.inst[2] = { X2, WPctxf, ctxf_b, HSs2f, nullptr, 0 };
    a.inst[3] = { X2, WPctxb, ctxb_b, HSs2b, nullptr, 1 };
    lstm_kernel<X_KXP, X_KP><<<dim3(B_ / 16, 4), 256, 0, stream>>>(a);
  }

  // 5) multi-perspective matching -> fp16 agg inputs
  match_kernel<<<dim3(B_ * S_, 2), 64, 0, stream>>>(HSs1f, HSs1b, HSs2f, HSs2b,
                                                    w_fw, w_bw, Xagg1, Xagg2);

  // 6) aggregation BiLSTMs (final states only)
  {
    LstmArgs a;
    a.inst[0] = { Xagg1, WPaggf, aggf_b, nullptr, h1f, 0 };
    a.inst[1] = { Xagg1, WPaggb, aggb_b, nullptr, h1b, 1 };
    a.inst[2] = { Xagg2, WPaggf, aggf_b, nullptr, h2f, 0 };
    a.inst[3] = { Xagg2, WPaggb, aggb_b, nullptr, h2b, 1 };
    lstm_kernel<A_KXP, A_KP><<<dim3(B_ / 16, 4), 256, 0, stream>>>(a);
  }

  // 7) prediction head
  pred_kernel<<<dim3(2), 256, 0, stream>>>(h1f, h1b, h2f, h2b, pred_W, pred_b, out);
}
